// PointGLEncoder_46832323396215
// MI455X (gfx1250) — matrix operand order repaired and checked by reading, not compiled, not run on hardware
//
#include <hip/hip_runtime.h>

// ---------------------------------------------------------------------------
// CDNA5 (gfx1250) PointGL encoder. Matmuls use V_WMMA_F32_16X16X32_BF16 with
// pre-packed bf16 operands so the inner loop is pure b128 loads + wmma.
// ---------------------------------------------------------------------------

typedef __attribute__((ext_vector_type(16))) __bf16 v16bf;
typedef __attribute__((ext_vector_type(8)))  float  v8f;

union PGFrag { v16bf v; uint4 q[2]; };

#define PG_B   16
#define PG_N0  2048
#define PG_K   32

// ---------------------------------------------------------------------------
// Embedding: xcat[b, 0:3, n] = xyz(raw);  xcat[b, 3:67, n] = relu(bn(W*xyz+b))
// ---------------------------------------------------------------------------
__global__ __launch_bounds__(256)
void pg_embed_kernel(const float* __restrict__ xyz, const float* __restrict__ W,
                     const float* __restrict__ bias, const float* __restrict__ s,
                     const float* __restrict__ t, float* __restrict__ xc,
                     int N, int total)
{
    int idx = blockIdx.x * blockDim.x + threadIdx.x;
    if (idx >= total) return;
    int b = idx / N, n = idx - b * N;
    const float* p = xyz + (size_t)idx * 3;
    float px = p[0], py = p[1], pz = p[2];
    float* out = xc + (size_t)b * 67 * N;
    out[(size_t)0 * N + n] = px;
    out[(size_t)1 * N + n] = py;
    out[(size_t)2 * N + n] = pz;
    #pragma unroll 4
    for (int o = 0; o < 64; ++o) {
        float y = W[o*3+0]*px + W[o*3+1]*py + W[o*3+2]*pz + bias[o];
        y = s[o]*y + t[o];
        out[(size_t)(3+o)*N + n] = fmaxf(y, 0.f);
    }
}

// ---------------------------------------------------------------------------
// Operand packing for WMMA: f32 -> bf16, zero-padded K to multiple of 32.
// ---------------------------------------------------------------------------
__global__ __launch_bounds__(256)
void pg_pack_w_kernel(const float* __restrict__ W, __bf16* __restrict__ Wb,
                      int Cin, int Cinp, int total)
{
    int idx = blockIdx.x * blockDim.x + threadIdx.x;
    if (idx >= total) return;
    int kp = idx % Cinp;
    int m  = idx / Cinp;
    float v = (kp < Cin) ? W[(size_t)m * Cin + kp] : 0.f;
    Wb[idx] = (__bf16)v;
}

// X [B,Cin,N] f32 -> Xt [B,N,Cinp] bf16 (transpose + pad)
__global__ __launch_bounds__(256)
void pg_pack_x_kernel(const float* __restrict__ X, __bf16* __restrict__ Xt,
                      int Cin, int Cinp, int N, int total)
{
    int idx = blockIdx.x * blockDim.x + threadIdx.x;
    if (idx >= total) return;
    int kp = idx % Cinp;
    int rest = idx / Cinp;
    int n = rest % N;
    int b = rest / N;
    float v = (kp < Cin) ? X[((size_t)b * Cin + kp) * N + n] : 0.f;
    Xt[idx] = (__bf16)v;
}

// ---------------------------------------------------------------------------
// Fused ConvBN GEMM on packed bf16 operands:
//   Y[b,m,n] = act( s[m]*( sum_k W[m,k]*X[b,k,n] + bias[m] ) + t[m] ) (+res)
// One wave32 computes a 16x32 tile (two 16x16 accumulators sharing the A
// fragment). Inner loop: 6 x global_load_b128 + 2 x v_wmma, no branches.
// grid = (N/128, Cout/16, B), block = 128 (4 waves).
// ---------------------------------------------------------------------------
__global__ __launch_bounds__(128)
void pg_convbn_wmma_kernel(const __bf16* __restrict__ Xt,   // [B, N, Cinp]
                           const __bf16* __restrict__ Wb,   // [Cout, Cinp]
                           const float* __restrict__ bias, const float* __restrict__ s,
                           const float* __restrict__ t, const float* __restrict__ res,
                           float* __restrict__ Y,
                           int Cinp, int Cout, int N, int relu)
{
    const int lane = threadIdx.x & 31;
    const int wv   = threadIdx.x >> 5;
    const int l15  = lane & 15;
    const int half = lane >> 4;                    // 0: lanes 0-15, 1: 16-31
    const int b    = blockIdx.z;
    const int m0   = blockIdx.y << 4;
    const int n0   = ((blockIdx.x << 2) + wv) << 5;  // 32 cols per wave

    const int mA = m0 + l15;
    // A fragment (16x32 bf16): lane row mA; K = 16*half + (0..15) contiguous
    const __bf16* Arow  = Wb + (size_t)mA * Cinp + 16 * half;
    // B fragments (32x16 bf16): lane col; K = 16*half + (0..15) contiguous
    const __bf16* Brow0 = Xt + ((size_t)b * N + n0 + l15) * Cinp + 16 * half;
    const __bf16* Brow1 = Brow0 + (size_t)16 * Cinp;

    v8f acc0 = {0.f,0.f,0.f,0.f,0.f,0.f,0.f,0.f};
    v8f acc1 = {0.f,0.f,0.f,0.f,0.f,0.f,0.f,0.f};

    #pragma unroll 2
    for (int k0 = 0; k0 < Cinp; k0 += 32) {
        PGFrag a, b0, b1;
        a.q[0]  = *(const uint4*)(Arow  + k0);
        a.q[1]  = *(const uint4*)(Arow  + k0 + 8);
        b0.q[0] = *(const uint4*)(Brow0 + k0);
        b0.q[1] = *(const uint4*)(Brow0 + k0 + 8);
        b1.q[0] = *(const uint4*)(Brow1 + k0);
        b1.q[1] = *(const uint4*)(Brow1 + k0 + 8);
        acc0 = __builtin_amdgcn_wmma_f32_16x16x32_bf16(
            false, a.v, false, b0.v, (short)0, acc0, false, false);
        acc1 = __builtin_amdgcn_wmma_f32_16x16x32_bf16(
            false, a.v, false, b1.v, (short)0, acc1, false, false);
    }

    // Epilogue: D element v of lane -> M = m0+v+8*half, N = col
    #pragma unroll
    for (int v = 0; v < 8; ++v) {
        int mr = m0 + v + (half << 3);
        float sc = s[mr], bi = bias[mr], tr = t[mr];
        size_t rowoff = ((size_t)b * Cout + mr) * N;

        float y0 = sc * (acc0[v] + bi) + tr;
        float y1 = sc * (acc1[v] + bi) + tr;
        if (relu) { y0 = fmaxf(y0, 0.f); y1 = fmaxf(y1, 0.f); }
        size_t o0 = rowoff + n0 + l15;
        size_t o1 = o0 + 16;
        if (res) { y0 += res[o0]; y1 += res[o1]; }
        Y[o0] = y0;
        Y[o1] = y1;
    }
}

// ---------------------------------------------------------------------------
// Furthest point sampling (CUDA-kernel semantics): one block per batch.
// dist[] kept resident in LDS; first-index tiebreak argmax (matches argmax).
// ---------------------------------------------------------------------------
__global__ __launch_bounds__(256)
void pg_fps_kernel(const float* __restrict__ xyz, int* __restrict__ fidx,
                   int N, int G)
{
    __shared__ float dist[PG_N0];
    __shared__ float rv[256];
    __shared__ int   ri[256];
    __shared__ int   s_last;

    const int b = blockIdx.x;
    const int tid = threadIdx.x;
    const float* P = xyz + (size_t)b * N * 3;

    for (int n = tid; n < N; n += 256) dist[n] = 1e10f;
    if (tid == 0) { fidx[b * G] = 0; s_last = 0; }
    __syncthreads();

    for (int g = 1; g < G; ++g) {
        int last = s_last;
        float lx = P[last*3+0], ly = P[last*3+1], lz = P[last*3+2];
        float bv = -1.f; int bi = 0;
        for (int n = tid; n < N; n += 256) {
            float dx = P[n*3+0]-lx, dy = P[n*3+1]-ly, dz = P[n*3+2]-lz;
            float d = dx*dx + dy*dy + dz*dz;
            d = fminf(dist[n], d);
            dist[n] = d;
            if (d > bv) { bv = d; bi = n; }
        }
        rv[tid] = bv; ri[tid] = bi;
        __syncthreads();
        for (int st = 128; st > 0; st >>= 1) {
            if (tid < st) {
                float ov = rv[tid+st]; int oi = ri[tid+st];
                if (ov > rv[tid] || (ov == rv[tid] && oi < ri[tid])) {
                    rv[tid] = ov; ri[tid] = oi;
                }
            }
            __syncthreads();
        }
        if (tid == 0) { s_last = ri[0]; fidx[b*G + g] = ri[0]; }
        __syncthreads();
    }
}

// new_xyz[b,g,:] = xyz[b, fidx[b,g], :]
__global__ __launch_bounds__(256)
void pg_gather_xyz_kernel(const float* __restrict__ xyz, const int* __restrict__ fidx,
                          float* __restrict__ nxyz, int N, int G, int total)
{
    int idx = blockIdx.x * blockDim.x + threadIdx.x;
    if (idx >= total) return;
    int b = idx / G;
    const float* src = xyz + ((size_t)b * N + fidx[idx]) * 3;
    float* dst = nxyz + (size_t)idx * 3;
    dst[0] = src[0]; dst[1] = src[1]; dst[2] = src[2];
}

// First-K-in-radius ball query, pad with first hit (center guarantees >=1 hit)
__global__ __launch_bounds__(256)
void pg_ballquery_kernel(const float* __restrict__ nxyz, const float* __restrict__ xyz,
                         int* __restrict__ nidx, int N, int G, float r2, int total)
{
    int idx = blockIdx.x * blockDim.x + threadIdx.x;
    if (idx >= total) return;
    int b = idx / G;
    const float* q = nxyz + (size_t)idx * 3;
    float qx = q[0], qy = q[1], qz = q[2];
    const float* P = xyz + (size_t)b * N * 3;
    int* out = nidx + (size_t)idx * PG_K;
    int cnt = 0, first = 0;
    for (int n = 0; n < N && cnt < PG_K; ++n) {
        float dx = P[n*3+0]-qx, dy = P[n*3+1]-qy, dz = P[n*3+2]-qz;
        if (dx*dx + dy*dy + dz*dz < r2) {
            if (cnt == 0) first = n;
            out[cnt++] = n;
        }
    }
    for (; cnt < PG_K; ++cnt) out[cnt] = first;
}

// Grouping: gather K neighbors of channel d, center by mean, affine, max over K.
// Output xn[b, 0:D, g] = pooled features, xn[b, D:D+3, g] = new_xyz.
__global__ __launch_bounds__(256)
void pg_group_kernel(const float* __restrict__ x, const int* __restrict__ nidx,
                     const float* __restrict__ nxyz, const float* __restrict__ alpha,
                     const float* __restrict__ beta, float* __restrict__ xn,
                     int D, int N, int G, int total)
{
    int idx = blockIdx.x * blockDim.x + threadIdx.x;
    if (idx >= total) return;
    int g = idx % G;
    int rest = idx / G;
    int d = rest % (D + 3);
    int b = rest / (D + 3);

    if (d >= D) {
        xn[idx] = nxyz[((size_t)b * G + g) * 3 + (d - D)];
        return;
    }
    const int* ni = nidx + ((size_t)b * G + g) * PG_K;
    const float* xd = x + ((size_t)b * D + d) * N;
    float vals[PG_K];
    float sum = 0.f;
    #pragma unroll
    for (int k = 0; k < PG_K; ++k) {
        float v = xd[ni[k]];
        vals[k] = v;
        sum += v;
    }
    float mean = sum * (1.f / PG_K);
    float a = alpha[d], bb = beta[d];
    float mx = -3.402823466e38f;
    #pragma unroll
    for (int k = 0; k < PG_K; ++k)
        mx = fmaxf(mx, a * (vals[k] - mean) + bb);
    xn[idx] = mx;
}

// Final adaptive max pool over points: out[b,c] = max_g x[b,c,g]
__global__ __launch_bounds__(256)
void pg_final_max_kernel(const float* __restrict__ x, float* __restrict__ out,
                         int C, int G, int total)
{
    int idx = blockIdx.x * blockDim.x + threadIdx.x;
    if (idx >= total) return;
    const float* xs = x + (size_t)idx * G;
    float mx = xs[0];
    for (int g = 1; g < G; ++g) mx = fmaxf(mx, xs[g]);
    out[idx] = mx;
}

// ---------------------------------------------------------------------------
// Host orchestration
// ---------------------------------------------------------------------------
struct PGConv { const float *W, *b, *s, *t; };
struct PGStage { PGConv tr; PGConv blk[2][2]; const float *alpha, *beta; };

extern "C" void kernel_launch(void* const* d_in, const int* in_sizes, int n_in,
                              void* d_out, int out_size, void* d_ws, size_t ws_size,
                              hipStream_t stream)
{
    (void)n_in; (void)out_size; (void)ws_size;
    const float radii[4] = {0.1f, 0.2f, 0.4f, 0.8f};

    // ---- parse inputs (handles insertion-order or sorted-key flattening) ----
    int p = 0;
    auto F = [&](int i) { return (const float*)d_in[i]; };
    PGConv emb; PGStage st[4];
    const float* xyz_in;
    const bool insertion = (in_sizes[0] == PG_B * PG_N0 * 3);

    auto takeConv = [&](PGConv& c) {
        c.W = F(p++); c.b = F(p++); c.s = F(p++); c.t = F(p++);
    };

    if (insertion) {
        xyz_in = F(p++);
        takeConv(emb);
        for (int i = 0; i < 4; ++i) {
            takeConv(st[i].tr);
            for (int j = 0; j < 2; ++j) { takeConv(st[i].blk[j][0]); takeConv(st[i].blk[j][1]); }
            st[i].alpha = F(p++); st[i].beta = F(p++);
        }
    } else { // jax tree_flatten (sorted dict keys): params first, xyz last
        takeConv(emb);
        for (int i = 0; i < 4; ++i) {
            st[i].alpha = F(p++); st[i].beta = F(p++);
            for (int j = 0; j < 2; ++j) { takeConv(st[i].blk[j][0]); takeConv(st[i].blk[j][1]); }
            takeConv(st[i].tr);
        }
        xyz_in = F(p++);
    }

    // ---- workspace carve ----
    const size_t XBUF = (size_t)PG_B * 262144;  // per-buffer floats
    float* bufs[3];
    bufs[0] = (float*)d_ws;
    bufs[1] = bufs[0] + XBUF;
    bufs[2] = bufs[1] + XBUF;
    float* xyzA = bufs[2] + XBUF;                         // B*2048*3 floats
    float* xyzB = xyzA + (size_t)PG_B * PG_N0 * 3;
    int*   fidx = (int*)(xyzB + (size_t)PG_B * PG_N0 * 3);
    int*   nidx = fidx + PG_B * 1024;                     // B*1024*32 ints
    __bf16* Xt  = (__bf16*)(nidx + PG_B * 1024 * PG_K);   // B*262144 bf16
    __bf16* Wb  = Xt + XBUF;                              // 1024*1024 bf16

    auto launch_conv = [&](const float* X, const PGConv& c, const float* res,
                           float* Y, int Cin, int Cout, int N, int relu) {
        const int Cinp = (Cin + 31) & ~31;
        // pack weights -> bf16 [Cout, Cinp]
        {
            int tot = Cout * Cinp;
            pg_pack_w_kernel<<<(tot + 255) / 256, 256, 0, stream>>>(
                c.W, Wb, Cin, Cinp, tot);
        }
        // pack + transpose activations -> bf16 [B, N, Cinp]
        {
            int tot = PG_B * N * Cinp;
            pg_pack_x_kernel<<<(tot + 255) / 256, 256, 0, stream>>>(
                X, Xt, Cin, Cinp, N, tot);
        }
        dim3 grid(N / 128, Cout / 16, PG_B);
        pg_convbn_wmma_kernel<<<grid, 128, 0, stream>>>(
            Xt, Wb, c.b, c.s, c.t, res, Y, Cinp, Cout, N, relu);
    };

    // ---- embedding + use_xyz concat -> bufs[0] [B, 67, 2048] ----
    {
        int total = PG_B * PG_N0;
        pg_embed_kernel<<<(total + 255) / 256, 256, 0, stream>>>(
            xyz_in, emb.W, emb.b, emb.s, emb.t, bufs[0], PG_N0, total);
    }

    int xi = 0;
    int Nin = PG_N0;
    int Cprev = 64;
    const float* cur_xyz = xyz_in;
    float* nxbuf[2] = {xyzA, xyzB};
    int xz = 0;

    for (int i = 0; i < 4; ++i) {
        const int Cin_t = Cprev + 3;
        const int Cout  = Cprev * 2;

        // transfer conv (relu)
        int ti = (xi + 1) % 3;
        launch_conv(bufs[xi], st[i].tr, nullptr, bufs[ti], Cin_t, Cout, Nin, 1);
        xi = ti;

        // 2 residual blocks: relu(conv) -> conv -> +x
        for (int j = 0; j < 2; ++j) {
            int a = (xi + 1) % 3, c = (xi + 2) % 3;
            launch_conv(bufs[xi], st[i].blk[j][0], nullptr, bufs[a], Cout, Cout, Nin, 1);
            launch_conv(bufs[a],  st[i].blk[j][1], bufs[xi], bufs[c], Cout, Cout, Nin, 0);
            xi = c;
        }

        // grouper
        const int G = Nin / 2;
        pg_fps_kernel<<<PG_B, 256, 0, stream>>>(cur_xyz, fidx, Nin, G);

        float* newxyz = nxbuf[xz]; xz ^= 1;
        int bg = PG_B * G;
        pg_gather_xyz_kernel<<<(bg + 255) / 256, 256, 0, stream>>>(
            cur_xyz, fidx, newxyz, Nin, G, bg);

        float r = radii[i];
        pg_ballquery_kernel<<<(bg + 255) / 256, 256, 0, stream>>>(
            newxyz, cur_xyz, nidx, Nin, G, r * r, bg);

        int oi = (xi + 1) % 3;
        int tot = PG_B * (Cout + 3) * G;
        pg_group_kernel<<<(tot + 255) / 256, 256, 0, stream>>>(
            bufs[xi], nidx, newxyz, st[i].alpha, st[i].beta, bufs[oi],
            Cout, Nin, G, tot);
        xi = oi;

        cur_xyz = newxyz;
        Nin = G;
        Cprev = Cout;
    }

    // final max pool over remaining points: [B, 1027, 128] -> [B, 1027]
    {
        int C = Cprev + 3;
        int total = PG_B * C;
        pg_final_max_kernel<<<(total + 255) / 256, 256, 0, stream>>>(
            bufs[xi], (float*)d_out, C, Nin, total);
    }
}
